// StudentTeacherVGAE_54030688584372
// MI455X (gfx1250) — matrix operand order repaired and checked by reading, not compiled, not run on hardware
//
#include <hip/hip_runtime.h>

// ---------------------------------------------------------------------------
// StudentTeacherVGAE on gfx1250 (MI455X, wave32, WMMA)
// Dense GEMMs -> v_wmma_f32_16x16x32_bf16 (bf16 A/B, f32 accumulate).
// The dominant sigmoid(z@z.T) decoder uses async global->LDS staging
// (global_load_async_to_lds_b128 / s_wait_asynccnt) and non-temporal stores.
// ---------------------------------------------------------------------------

typedef __attribute__((ext_vector_type(16))) __bf16 v16bf;
typedef __attribute__((ext_vector_type(8)))  __bf16 v8bf;
typedef __attribute__((ext_vector_type(8)))  float  v8f;

#define NN    12288
#define EE    393216
#define FEATD 256
#define HD    256
#define LATD  64
#define TLATD 128
#define THIDD 512
#define EPSV  1e-5f

// ------------------------------ WMMA GEMM ----------------------------------
// C[M x Ncols] = act( A[M x K] * B + bias ), B is [K x Ncols] (transB=0) or
// [Ncols x K] (transB=1). Block tile 128x64, K-step 32, 8 waves, each wave
// computes a 32x32 quad of 16x16x32 bf16 WMMA tiles.
#define TM 128
#define TN 64
#define TK 32
#define LSTRIDE 40   // bf16 elements per LDS row (32 + 8 pad, keeps 16B align)

__launch_bounds__(256, 2)
__global__ void gemm_wmma_bf16(const float* __restrict__ A, int lda,
                               const float* __restrict__ B, int ldb, int transB,
                               const float* __restrict__ bias,
                               float* __restrict__ C, size_t ldc,
                               int K, int act)
{
    __shared__ __bf16 As[TM * LSTRIDE];
    __shared__ __bf16 Bs[TN * LSTRIDE];

    const int tid  = threadIdx.x;
    const int lane = tid & 31;
    const int wave = tid >> 5;
    const long bm  = (long)blockIdx.x * TM;
    const long bn  = (long)blockIdx.y * TN;
    const int  wm  = (wave & 3) * 32;
    const int  wn  = (wave >> 2) * 32;

    const int l15  = lane & 15;
    const int hsel = lane >> 4;

    v8f acc[2][2];
    #pragma unroll
    for (int mi = 0; mi < 2; ++mi)
        #pragma unroll
        for (int ni = 0; ni < 2; ++ni)
            acc[mi][ni] = (v8f){0.f,0.f,0.f,0.f,0.f,0.f,0.f,0.f};

    for (int k0 = 0; k0 < K; k0 += TK) {
        {   // stage A tile (128 x 32 fp32 -> bf16)
            const int r  = tid >> 1;
            const int cb = (tid & 1) * 16;
            const float* src = A + (bm + r) * (long)lda + (k0 + cb);
            __bf16* d = &As[r * LSTRIDE + cb];
            #pragma unroll
            for (int i = 0; i < 16; i += 4) {
                float4 v = *(const float4*)(src + i);
                d[i + 0] = (__bf16)v.x;
                d[i + 1] = (__bf16)v.y;
                d[i + 2] = (__bf16)v.z;
                d[i + 3] = (__bf16)v.w;
            }
            if (k0 + TK < K) __builtin_prefetch(src + TK, 0, 3);
        }
        {   // stage B tile transposed into Bs[n][k]
            const int n  = tid >> 2;
            const int kb = (tid & 3) * 8;
            __bf16* d = &Bs[n * LSTRIDE + kb];
            if (transB) {
                const float* src = B + (bn + n) * (long)ldb + (k0 + kb);
                float4 v0 = *(const float4*)(src);
                float4 v1 = *(const float4*)(src + 4);
                d[0] = (__bf16)v0.x; d[1] = (__bf16)v0.y;
                d[2] = (__bf16)v0.z; d[3] = (__bf16)v0.w;
                d[4] = (__bf16)v1.x; d[5] = (__bf16)v1.y;
                d[6] = (__bf16)v1.z; d[7] = (__bf16)v1.w;
                if (k0 + TK < K) __builtin_prefetch(src + TK, 0, 3);
            } else {
                const float* src = B + (long)(k0 + kb) * ldb + (bn + n);
                #pragma unroll
                for (int i = 0; i < 8; ++i)
                    d[i] = (__bf16)src[(long)i * ldb];
            }
        }
        __syncthreads();

        v16bf afrag[2], bfrag[2];
        #pragma unroll
        for (int mi = 0; mi < 2; ++mi) {
            const __bf16* base = &As[(wm + mi * 16 + l15) * LSTRIDE];
            v8bf lo = *(const v8bf*)(base + hsel * 8);
            v8bf hi = *(const v8bf*)(base + 16 + hsel * 8);
            afrag[mi] = __builtin_shufflevector(lo, hi,
                0,1,2,3,4,5,6,7,8,9,10,11,12,13,14,15);
        }
        #pragma unroll
        for (int ni = 0; ni < 2; ++ni) {
            const __bf16* base = &Bs[(wn + ni * 16 + l15) * LSTRIDE];
            v8bf lo = *(const v8bf*)(base + hsel * 8);
            v8bf hi = *(const v8bf*)(base + 16 + hsel * 8);
            bfrag[ni] = __builtin_shufflevector(lo, hi,
                0,1,2,3,4,5,6,7,8,9,10,11,12,13,14,15);
        }

        #pragma unroll
        for (int mi = 0; mi < 2; ++mi)
            #pragma unroll
            for (int ni = 0; ni < 2; ++ni)
                acc[mi][ni] = __builtin_amdgcn_wmma_f32_16x16x32_bf16(
                    false, afrag[mi], false, bfrag[ni],
                    (short)0, acc[mi][ni], false, false);

        __syncthreads();
    }

    #pragma unroll
    for (int mi = 0; mi < 2; ++mi) {
        #pragma unroll
        for (int ni = 0; ni < 2; ++ni) {
            const long col  = bn + wn + ni * 16 + l15;
            const long row0 = bm + wm + mi * 16 + hsel * 8;
            const float bv  = bias ? bias[col] : 0.0f;
            #pragma unroll
            for (int i = 0; i < 8; ++i) {
                float v = acc[mi][ni][i] + bv;
                if (act == 1)      v = fmaxf(v, 0.0f);
                else if (act == 2) v = 1.0f / (1.0f + __expf(-v));
                C[(size_t)(row0 + i) * ldc + col] = v;
            }
        }
    }
}

// --------------- specialized adj = sigmoid(z @ z.T) decoder ----------------
// z pre-converted to bf16 [N x 64] densely packed. Block tile 128x128, whole
// K=64 resident in LDS, staged with async global->LDS copies (ASYNCcnt),
// 16 WMMAs per wave, sigmoid + non-temporal (NT) stores for the 604MB output.
#define ALD 72   // bf16 row stride in LDS (64 + 8 pad = 144B, 16B aligned)

__device__ __forceinline__ void async_cp16(unsigned lds_off, const __bf16* gptr) {
    unsigned long long ga = (unsigned long long)(size_t)gptr;
    asm volatile("global_load_async_to_lds_b128 %0, %1, off"
                 :: "v"(lds_off), "v"(ga) : "memory");
}

__launch_bounds__(256)
__global__ void adj_wmma_async(const __bf16* __restrict__ zb,
                               float* __restrict__ adj)
{
    __shared__ __bf16 As[128 * ALD];
    __shared__ __bf16 Bs[128 * ALD];

    const int tid  = threadIdx.x;
    const int lane = tid & 31;
    const int wave = tid >> 5;
    const long bm  = (long)blockIdx.x * 128;
    const long bn  = (long)blockIdx.y * 128;
    const int  wm  = (wave & 3) * 32;   // 4 row groups of 32
    const int  wn  = (wave >> 2) * 64;  // 2 col groups of 64
    const int  l15  = lane & 15;
    const int  hsel = lane >> 4;

    // ---- async stage: 128x64 bf16 = 1024 x 16B chunks per tile ----
    const unsigned asb = (unsigned)(size_t)&As[0];
    const unsigned bsb = (unsigned)(size_t)&Bs[0];
    const __bf16* ga = zb + (size_t)bm * LATD;   // tile rows contiguous
    const __bf16* gb = zb + (size_t)bn * LATD;
    #pragma unroll
    for (int i = 0; i < 4; ++i) {
        const int idx = tid + i * 256;       // chunk 0..1023
        const int r = idx >> 3;              // row 0..127
        const int c = idx & 7;               // 16B chunk within row
        async_cp16(asb + r * (ALD * 2) + c * 16, ga + r * LATD + c * 8);
        async_cp16(bsb + r * (ALD * 2) + c * 16, gb + r * LATD + c * 8);
    }
    asm volatile("s_wait_asynccnt 0" ::: "memory");
    __syncthreads();

    // ---- compute: K = 64 -> two 16x16x32 WMMA chunks ----
    v8f acc[2][4];
    #pragma unroll
    for (int mi = 0; mi < 2; ++mi)
        #pragma unroll
        for (int ni = 0; ni < 4; ++ni)
            acc[mi][ni] = (v8f){0.f,0.f,0.f,0.f,0.f,0.f,0.f,0.f};

    #pragma unroll
    for (int k0 = 0; k0 < LATD; k0 += 32) {
        v16bf afrag[2], bfrag[4];
        #pragma unroll
        for (int mi = 0; mi < 2; ++mi) {
            const __bf16* base = &As[(wm + mi * 16 + l15) * ALD] + k0;
            v8bf lo = *(const v8bf*)(base + hsel * 8);
            v8bf hi = *(const v8bf*)(base + 16 + hsel * 8);
            afrag[mi] = __builtin_shufflevector(lo, hi,
                0,1,2,3,4,5,6,7,8,9,10,11,12,13,14,15);
        }
        #pragma unroll
        for (int ni = 0; ni < 4; ++ni) {
            const __bf16* base = &Bs[(wn + ni * 16 + l15) * ALD] + k0;
            v8bf lo = *(const v8bf*)(base + hsel * 8);
            v8bf hi = *(const v8bf*)(base + 16 + hsel * 8);
            bfrag[ni] = __builtin_shufflevector(lo, hi,
                0,1,2,3,4,5,6,7,8,9,10,11,12,13,14,15);
        }
        #pragma unroll
        for (int mi = 0; mi < 2; ++mi)
            #pragma unroll
            for (int ni = 0; ni < 4; ++ni)
                acc[mi][ni] = __builtin_amdgcn_wmma_f32_16x16x32_bf16(
                    false, afrag[mi], false, bfrag[ni],
                    (short)0, acc[mi][ni], false, false);
    }

    // ---- epilogue: sigmoid, stream out with non-temporal stores ----
    #pragma unroll
    for (int mi = 0; mi < 2; ++mi) {
        #pragma unroll
        for (int ni = 0; ni < 4; ++ni) {
            const long col  = bn + wn + ni * 16 + l15;
            const long row0 = bm + wm + mi * 16 + hsel * 8;
            #pragma unroll
            for (int i = 0; i < 8; ++i) {
                float v = 1.0f / (1.0f + __expf(-acc[mi][ni][i]));
                __builtin_nontemporal_store(v, &adj[(size_t)(row0 + i) * NN + col]);
            }
        }
    }
}

// --------------------------- small helper kernels --------------------------

__global__ void k_fill(float* __restrict__ p, float v, long n) {
    long i = (long)blockIdx.x * blockDim.x + threadIdx.x;
    if (i < n) p[i] = v;
}

__global__ void k_cvt_bf16(const float* __restrict__ s, __bf16* __restrict__ d, long n) {
    long i = (long)blockIdx.x * blockDim.x + threadIdx.x;
    if (i < n) d[i] = (__bf16)s[i];
}

__global__ void k_deg_accum(const int* __restrict__ dst, float* __restrict__ deg) {
    int e = blockIdx.x * blockDim.x + threadIdx.x;
    if (e < EE) atomicAdd(&deg[dst[e]], 1.0f);
}

__global__ void k_rsqrt(const float* __restrict__ deg, float* __restrict__ dinv) {
    int i = blockIdx.x * blockDim.x + threadIdx.x;
    if (i < NN) dinv[i] = rsqrtf(deg[i]);
}

__global__ void k_edge_agg(const int* __restrict__ src, const int* __restrict__ dst,
                           const float* __restrict__ dinv,
                           const float* __restrict__ h, float* __restrict__ agg,
                           int C)
{
    const int e = blockIdx.x;
    const int c = threadIdx.x;
    const int s = src[e];
    const int d = dst[e];
    const float coef = dinv[s] * dinv[d];
    atomicAdd(&agg[(long)d * C + c], h[(long)s * C + c] * coef);
}

__global__ void k_gcn_finalize(float* __restrict__ agg, const float* __restrict__ h,
                               const float* __restrict__ dinv,
                               const float* __restrict__ b, int C, long total)
{
    long idx = (long)blockIdx.x * blockDim.x + threadIdx.x;
    if (idx >= total) return;
    const long r = idx / C;
    const int  c = (int)(idx - r * C);
    const float di = dinv[r];
    agg[idx] = agg[idx] + h[idx] * di * di + b[c];
}

__global__ void k_colstats(const float* __restrict__ X, int M, int C,
                           float* __restrict__ mean, float* __restrict__ rstd)
{
    const int c = blockIdx.x;
    float s = 0.f, s2 = 0.f;
    for (int r = threadIdx.x; r < M; r += blockDim.x) {
        const float v = X[(long)r * C + c];
        s  += v;
        s2 += v * v;
    }
    __shared__ float sh[256], sh2[256];
    sh[threadIdx.x] = s; sh2[threadIdx.x] = s2;
    __syncthreads();
    for (int off = 128; off > 0; off >>= 1) {
        if (threadIdx.x < off) {
            sh[threadIdx.x]  += sh[threadIdx.x + off];
            sh2[threadIdx.x] += sh2[threadIdx.x + off];
        }
        __syncthreads();
    }
    if (threadIdx.x == 0) {
        const float m   = sh[0] / (float)M;
        const float var = sh2[0] / (float)M - m * m;
        mean[c] = m;
        rstd[c] = rsqrtf(var + EPSV);
    }
}

__global__ void k_bn_act(float* __restrict__ X, long total, int C,
                         const float* __restrict__ mean, const float* __restrict__ rstd,
                         const float* __restrict__ g, const float* __restrict__ be,
                         int relu)
{
    long idx = (long)blockIdx.x * blockDim.x + threadIdx.x;
    if (idx >= total) return;
    const int c = (int)(idx % C);
    float v = (X[idx] - mean[c]) * rstd[c] * g[c] + be[c];
    if (relu) v = fmaxf(v, 0.0f);
    X[idx] = v;
}

// ------------------------------- launcher ----------------------------------

extern "C" void kernel_launch(void* const* d_in, const int* in_sizes, int n_in,
                              void* d_out, int out_size, void* d_ws, size_t ws_size,
                              hipStream_t stream) {
    (void)in_sizes; (void)n_in; (void)out_size; (void)ws_size;

    const float* x    = (const float*)d_in[0];
    const int*   edge = (const int*)  d_in[1];
    const float* W1   = (const float*)d_in[2];
    const float* b1   = (const float*)d_in[3];
    const float* g1   = (const float*)d_in[4];
    const float* be1  = (const float*)d_in[5];
    const float* W2   = (const float*)d_in[6];
    const float* b2   = (const float*)d_in[7];
    const float* g2   = (const float*)d_in[8];
    const float* be2  = (const float*)d_in[9];
    const float* Wmu  = (const float*)d_in[10];
    const float* bmu  = (const float*)d_in[11];
    const float* Wlv  = (const float*)d_in[12];
    const float* blv  = (const float*)d_in[13];
    const float* Wp   = (const float*)d_in[14];
    const float* bp   = (const float*)d_in[15];
    const float* gp   = (const float*)d_in[16];
    const float* bep  = (const float*)d_in[17];
    const float* Tw1  = (const float*)d_in[18];
    const float* Tb1  = (const float*)d_in[19];
    const float* Tw2  = (const float*)d_in[20];
    const float* Tb2  = (const float*)d_in[21];

    const int* esrc = edge;
    const int* edst = edge + EE;

    // d_out layout: adj [N,N] | x_recon [N,FEAT] | mu [N,LAT] | logvar [N,LAT]
    float* adj  = (float*)d_out;
    float* xrec = adj  + (size_t)NN * NN;
    float* mu   = xrec + (size_t)NN * FEATD;
    float* lv   = mu   + (size_t)NN * LATD;

    // workspace layout (floats)
    float* deg  = (float*)d_ws;
    float* dinv = deg  + NN;
    float* buf1 = dinv + NN;                   // N x 512 scratch
    float* buf2 = buf1 + (size_t)NN * THIDD;   // N x 512 scratch
    float* mean = buf2 + (size_t)NN * THIDD;   // 512
    float* rstd = mean + 512;                  // 512
    __bf16* zb  = (__bf16*)(rstd + 512);       // N x 64 bf16

    const long NH = (long)NN * HD;

    // --- degrees & symmetric normalization ---
    k_fill<<<NN / 256, 256, 0, stream>>>(deg, 1.0f, NN);   // +1 self loop
    k_deg_accum<<<EE / 256, 256, 0, stream>>>(edst, deg);
    k_rsqrt<<<NN / 256, 256, 0, stream>>>(deg, dinv);

    // --- GCN layer 1 ---
    dim3 gHH(NN / TM, HD / TN);
    gemm_wmma_bf16<<<gHH, 256, 0, stream>>>(x, FEATD, W1, HD, 0, nullptr,
                                            buf1, HD, FEATD, 0);
    k_fill<<<(NH + 255) / 256, 256, 0, stream>>>(buf2, 0.0f, NH);
    k_edge_agg<<<EE, HD, 0, stream>>>(esrc, edst, dinv, buf1, buf2, HD);
    k_gcn_finalize<<<(NH + 255) / 256, 256, 0, stream>>>(buf2, buf1, dinv, b1, HD, NH);
    k_colstats<<<HD, 256, 0, stream>>>(buf2, NN, HD, mean, rstd);
    k_bn_act<<<(NH + 255) / 256, 256, 0, stream>>>(buf2, NH, HD, mean, rstd, g1, be1, 1);

    // --- GCN layer 2 ---
    gemm_wmma_bf16<<<gHH, 256, 0, stream>>>(buf2, HD, W2, HD, 0, nullptr,
                                            buf1, HD, HD, 0);
    k_fill<<<(NH + 255) / 256, 256, 0, stream>>>(buf2, 0.0f, NH);
    k_edge_agg<<<EE, HD, 0, stream>>>(esrc, edst, dinv, buf1, buf2, HD);
    k_gcn_finalize<<<(NH + 255) / 256, 256, 0, stream>>>(buf2, buf1, dinv, b2, HD, NH);
    k_colstats<<<HD, 256, 0, stream>>>(buf2, NN, HD, mean, rstd);
    k_bn_act<<<(NH + 255) / 256, 256, 0, stream>>>(buf2, NH, HD, mean, rstd, g2, be2, 1);

    // --- mu / logvar heads (z == mu in eval mode) ---
    dim3 gLAT(NN / TM, LATD / TN);
    gemm_wmma_bf16<<<gLAT, 256, 0, stream>>>(buf2, HD, Wmu, LATD, 0, bmu,
                                             mu, LATD, HD, 0);
    gemm_wmma_bf16<<<gLAT, 256, 0, stream>>>(buf2, HD, Wlv, LATD, 0, blv,
                                             lv, LATD, HD, 0);

    // --- inner-product decoder: adj = sigmoid(z @ z.T) ---
    const long NZ = (long)NN * LATD;
    k_cvt_bf16<<<(NZ + 255) / 256, 256, 0, stream>>>(mu, zb, NZ);
    dim3 gADJ(NN / 128, NN / 128);
    adj_wmma_async<<<gADJ, 256, 0, stream>>>(zb, adj);

    // --- latent projection + BN (no activation) ---
    const long NP = (long)NN * TLATD;
    dim3 gP(NN / TM, TLATD / TN);
    gemm_wmma_bf16<<<gP, 256, 0, stream>>>(mu, LATD, Wp, TLATD, 0, bp,
                                           buf1, TLATD, LATD, 0);
    k_colstats<<<TLATD, 256, 0, stream>>>(buf1, NN, TLATD, mean, rstd);
    k_bn_act<<<(NP + 255) / 256, 256, 0, stream>>>(buf1, NP, TLATD, mean, rstd,
                                                   gp, bep, 0);

    // --- frozen teacher decoder MLP ---
    dim3 gT1(NN / TM, THIDD / TN);
    gemm_wmma_bf16<<<gT1, 256, 0, stream>>>(buf1, TLATD, Tw1, THIDD, 0, Tb1,
                                            buf2, THIDD, TLATD, 1);
    dim3 gT2(NN / TM, FEATD / TN);
    gemm_wmma_bf16<<<gT2, 256, 0, stream>>>(buf2, THIDD, Tw2, FEATD, 0, Tb2,
                                            xrec, FEATD, THIDD, 0);
}
